// MultiHeadInfiniAttention_49692771614932
// MI455X (gfx1250) — matrix operand order repaired and checked by reading, not compiled, not run on hardware
//
#include <hip/hip_runtime.h>

// ---------------- types ----------------
typedef __attribute__((ext_vector_type(16))) __bf16 v16bf;
typedef __attribute__((ext_vector_type(8)))  __bf16 bf16x8;
typedef __attribute__((ext_vector_type(8)))  float  v8f;
typedef __attribute__((ext_vector_type(2)))  float  v2f;
typedef __attribute__((ext_vector_type(4)))  unsigned int u32x4;
typedef __attribute__((ext_vector_type(8)))  unsigned int u32x8;

union V16U { v16bf v; bf16x8 h[2]; };

#define BATCH 2
#define SEQ   4096
#define DIN   1024
#define NH    8
#define DK    128
#define DV    128
#define LSEG  512
#define NSEG  8
#define NQK   1024   // NH*DK
#define NROWS 8192   // BATCH*SEQ
#define SPAD  516    // LDS row stride (floats); 8*516 % 64 = 32 -> no bank clash across halves

__device__ inline v8f zero8() {
  v8f z;
#pragma unroll
  for (int i = 0; i < 8; ++i) z[i] = 0.f;
  return z;
}

// ---------------- one-time conversions ----------------
__global__ void conv_x_kernel(const float* __restrict__ x, __bf16* __restrict__ xbf, int n) {
  int i = blockIdx.x * blockDim.x + threadIdx.x;
  if (i < n) xbf[i] = (__bf16)x[i];
}

// W [D,N] f32 -> Wt [mat][N][D] bf16 (so B-fragments read contiguous k)
__global__ void conv_w_kernel(const float* __restrict__ Wq, const float* __restrict__ Wk,
                              const float* __restrict__ Wv, __bf16* __restrict__ wt) {
  int i = blockIdx.x * blockDim.x + threadIdx.x;
  int mat = blockIdx.z;
  if (i >= DIN * NQK) return;
  const float* W = mat == 0 ? Wq : (mat == 1 ? Wk : Wv);
  int nn = i >> 10;         // output column 0..1023
  int d  = i & 1023;        // input dim
  wt[(size_t)mat * (DIN * NQK) + (size_t)nn * DIN + d] = (__bf16)W[(size_t)d * NQK + nn];
}

// ---------------- projection GEMM, 4x2 register tiling (spill-free) ----------------
// Wave computes a 64x32 output block: 8 WMMA per K-step off 4 A-frags + 2 B-frags.
// Single base pointer per matrix; tile-row offsets are compile-time immediates so
// the compiler keeps addressing in SALU + 24-bit instruction offsets (no spills).
__global__ void proj_kernel(const __bf16* __restrict__ xbf, const __bf16* __restrict__ wt,
                            const float* __restrict__ bq, const float* __restrict__ bk,
                            const float* __restrict__ bv,
                            float* __restrict__ Q, float* __restrict__ K, float* __restrict__ V) {
  int nt   = blockIdx.x;          // 0..31  (32-col blocks)
  int mt   = blockIdx.y;          // 0..127 (64-row blocks)
  int mat  = blockIdx.z;          // 0..2
  int lane = threadIdx.x;         // one wave per block
  int half = lane >> 4, nlow = lane & 15;

  const __bf16* w    = wt + (size_t)mat * (DIN * NQK);
  const float*  bias = mat == 0 ? bq : (mat == 1 ? bk : bv);
  float*        out  = mat == 0 ? Q : (mat == 1 ? K : V);

  const __bf16* abase = xbf + (size_t)(mt * 64 + nlow) * DIN;
  const __bf16* bbase = w   + (size_t)(nt * 32 + nlow) * DIN;

  v8f c[4][2];
#pragma unroll
  for (int mi = 0; mi < 4; ++mi)
#pragma unroll
    for (int ni = 0; ni < 2; ++ni) c[mi][ni] = zero8();

  for (int kb = 0; kb < DIN / 32; ++kb) {
    V16U a[4], b[2];
    int ka  = kb * 32 + half * 8;       // A: two 8-elem runs (k, k+16)
    int kbi = kb * 32 + half * 8;       // B: two 8-elem runs (k, k+16)
#pragma unroll
    for (int mi = 0; mi < 4; ++mi) {
      a[mi].h[0] = *(const bf16x8*)(abase + mi * 16 * DIN + ka);
      a[mi].h[1] = *(const bf16x8*)(abase + mi * 16 * DIN + ka + 16);
    }
#pragma unroll
    for (int ni = 0; ni < 2; ++ni) {
      b[ni].h[0] = *(const bf16x8*)(bbase + ni * 16 * DIN + kbi);
      b[ni].h[1] = *(const bf16x8*)(bbase + ni * 16 * DIN + kbi + 16);
    }
#pragma unroll
    for (int mi = 0; mi < 4; ++mi)
#pragma unroll
      for (int ni = 0; ni < 2; ++ni)
        c[mi][ni] = __builtin_amdgcn_wmma_f32_16x16x32_bf16(false, a[mi].v, false, b[ni].v,
                                                            (short)0, c[mi][ni], false, false);
  }
  float* obase = out + (size_t)(mt * 64 + 8 * half) * NQK + nt * 32 + nlow;
#pragma unroll
  for (int ni = 0; ni < 2; ++ni) {
    float bn = bias[nt * 32 + ni * 16 + nlow];
#pragma unroll
    for (int mi = 0; mi < 4; ++mi) {
#pragma unroll
      for (int r = 0; r < 8; ++r)
        obase[(size_t)(mi * 16 + r) * NQK + ni * 16] = c[mi][ni][r] + bn;
    }
  }
}

// ---------------- per-segment phi + transposed staging ----------------
__global__ void phi_kernel(const float* __restrict__ Q, const float* __restrict__ K,
                           const float* __restrict__ V, int seg,
                           float* __restrict__ sq, float* __restrict__ sk,
                           __bf16* __restrict__ qbf, __bf16* __restrict__ kbf,
                           __bf16* __restrict__ vT, __bf16* __restrict__ skT) {
  int idx = blockIdx.x * blockDim.x + threadIdx.x;   // < 16*512*128
  int bh  = idx >> 16;
  int rem = idx & 65535;
  int l = rem >> 7, d = rem & 127;
  int b = bh >> 3, h = bh & 7;
  size_t srow = (size_t)(b * SEQ + seg * LSEG + l) * NQK + h * DK + d;
  float qv = Q[srow], kv = K[srow], vv = V[srow];
  float sqv = qv > 0.f ? qv + 1.f : __expf(qv);      // elu(x)+1
  float skv = kv > 0.f ? kv + 1.f : __expf(kv);
  size_t base = (size_t)bh * 65536;
  sq[base + rem] = sqv;
  sk[base + rem] = skv;
  qbf[base + rem] = (__bf16)qv;
  kbf[base + rem] = (__bf16)kv;
  vT [base + (size_t)d * LSEG + l] = (__bf16)vv;
  skT[base + (size_t)d * LSEG + l] = (__bf16)skv;
}

// ---------------- retrieval + delta (f32 WMMA 16x16x4, K=128) ----------------
__global__ void retrieve_kernel(const float* __restrict__ sq, const float* __restrict__ sk,
                                const float* __restrict__ V,
                                const float* __restrict__ M, const float* __restrict__ z,
                                int seg, float* __restrict__ amem, __bf16* __restrict__ vdT) {
  int lt = blockIdx.x * 4 + (threadIdx.x >> 5);   // 0..31 (16-row tile of L)
  int bh = blockIdx.y;                            // 0..15
  int lane = threadIdx.x & 31;
  int half = lane >> 4, nlow = lane & 15;
  int b = bh >> 3, h = bh & 7;
  const float* Mb = M + (size_t)bh * (DK * DV);
  const float* zb = z + bh * DK;

  for (int phase = 0; phase < 2; ++phase) {
    const float* src = (phase ? sk : sq) + (size_t)bh * 65536;
    v8f c[8];
#pragma unroll
    for (int t = 0; t < 8; ++t) c[t] = zero8();

    const float* arow = src + (size_t)(lt * 16 + nlow) * DK;
    for (int kb = 0; kb < DK / 4; ++kb) {
      int k0 = kb * 4 + half * 2;       // f32 A 16x4: lane half -> k pair {2h, 2h+1}
      v2f a;
      a.x = arow[k0];
      a.y = arow[k0 + 1];
#pragma unroll
      for (int ntile = 0; ntile < 8; ++ntile) {
        v2f bv2;
        bv2.x = Mb[(size_t)k0 * DV + ntile * 16 + nlow];
        bv2.y = Mb[(size_t)(k0 + 1) * DV + ntile * 16 + nlow];
        c[ntile] = __builtin_amdgcn_wmma_f32_16x16x4_f32(false, a, false, bv2, (short)0,
                                                         c[ntile], false, false);
      }
    }
    // denominator for row (lt*16 + nlow): split across halves, combine via xor-16
    float den;
    {
      const float* sr = src + (size_t)(lt * 16 + nlow) * DK;
      float acc = 0.f;
      for (int d2 = half; d2 < DK; d2 += 2) acc += sr[d2] * zb[d2];
      acc += __shfl_xor(acc, 16);
      den = acc + 1e-6f;
    }
    if (phase == 0) {
      float* am = amem + (size_t)bh * 65536;
#pragma unroll
      for (int r = 0; r < 8; ++r) {
        int m = r + 8 * half;
        float dm = __shfl(den, m);      // stats live at lane == row index
        int l = lt * 16 + m;
#pragma unroll
        for (int ntile = 0; ntile < 8; ++ntile)
          am[(size_t)l * DK + ntile * 16 + nlow] = c[ntile][r] / dm;
      }
    } else {
      __bf16* vd = vdT + (size_t)bh * 65536;
#pragma unroll
      for (int r = 0; r < 8; ++r) {
        int m = r + 8 * half;
        float dm = __shfl(den, m);
        int l = lt * 16 + m;
        size_t srow = (size_t)(b * SEQ + seg * LSEG + l) * NQK + h * DK;
#pragma unroll
        for (int ntile = 0; ntile < 8; ++ntile) {
          int e = ntile * 16 + nlow;
          float vdel = V[srow + e] - c[ntile][r] / dm;
          vd[(size_t)e * LSEG + l] = (__bf16)vdel;   // store transposed for B-fragments
        }
      }
    }
  }
}

// ---------------- M += sk^T @ v_delta (bf16 WMMA, K=512) with TDM LDS staging ----------------
// Both 128KB panels are DMAd into LDS by the Tensor Data Mover (tensor_load_to_lds),
// then every wave reads fragments from LDS instead of re-reading L2 8x per panel row.
__global__ void update_kernel(const __bf16* __restrict__ skT, const __bf16* __restrict__ vdT,
                              const float* __restrict__ sk,
                              float* __restrict__ M, float* __restrict__ z) {
  extern __shared__ char lds[];
  __bf16* skl = (__bf16*)lds;                 // [128][512] bf16, LDS offset 0
  __bf16* vdl = (__bf16*)(lds + 131072);      // [128][512] bf16, LDS offset 128KB
  int bh = blockIdx.x;
  int tid = threadIdx.x;
  int wave = tid >> 5, lane = tid & 31;
  int half = lane >> 4, nlow = lane & 15;

  if (wave == 0) {
    // D# group0: count=1 | lds_addr | global_addr(57b) | type=2
    // D# group1: data_size=2B, tensor_dim0=512, tensor_dim1=128, tile 512x128, stride0=512
    unsigned long long ga = (unsigned long long)(skT + (size_t)bh * 65536);
    u32x4 g0;
    u32x8 g1;
    g0[0] = 1u;
    g0[1] = 0u;                                  // LDS byte address
    g0[2] = (unsigned)ga;
    g0[3] = (unsigned)((ga >> 32) & 0x01FFFFFFu) | (2u << 30);
    g1[0] = 1u << 16;                            // data_size=1 (2 bytes)
    g1[1] = 512u << 16;                          // tensor_dim0 = 512
    g1[2] = 128u << 16;                          // tensor_dim1 = 128
    g1[3] = 512u << 16;                          // tile_dim0 = 512
    g1[4] = 128u;                                // tile_dim1 = 128 (tile_dim2 = 0)
    g1[5] = 512u;                                // tensor_dim0_stride = 512
    g1[6] = 0u;
    g1[7] = 0u;
    asm volatile("tensor_load_to_lds %0, %1" :: "s"(g0), "s"(g1) : "memory");
    unsigned long long gb = (unsigned long long)(vdT + (size_t)bh * 65536);
    g0[1] = 131072u;
    g0[2] = (unsigned)gb;
    g0[3] = (unsigned)((gb >> 32) & 0x01FFFFFFu) | (2u << 30);
    asm volatile("tensor_load_to_lds %0, %1" :: "s"(g0), "s"(g1) : "memory");
    __builtin_amdgcn_s_wait_tensorcnt(0);
  }
  __syncthreads();

  for (int t = wave * 4; t < wave * 4 + 4; ++t) {   // 64 tiles / 16 waves
    int mt = t >> 3, nt = t & 7;
    v8f c;
#pragma unroll
    for (int r = 0; r < 8; ++r)
      c[r] = M[(size_t)bh * (DK * DV) + (size_t)(mt * 16 + r + 8 * half) * DV + nt * 16 + nlow];
    const __bf16* arow = skl + (size_t)(mt * 16 + nlow) * LSEG;
    const __bf16* brow = vdl + (size_t)(nt * 16 + nlow) * LSEG;
    for (int lb = 0; lb < LSEG; lb += 32) {
      V16U a, b2;
      int ka = lb + half * 8;
      a.h[0] = *(const bf16x8*)(arow + ka);
      a.h[1] = *(const bf16x8*)(arow + ka + 16);
      int kb2 = lb + half * 8;
      b2.h[0] = *(const bf16x8*)(brow + kb2);
      b2.h[1] = *(const bf16x8*)(brow + kb2 + 16);
      c = __builtin_amdgcn_wmma_f32_16x16x32_bf16(false, a.v, false, b2.v, (short)0, c, false, false);
    }
#pragma unroll
    for (int r = 0; r < 8; ++r)
      M[(size_t)bh * (DK * DV) + (size_t)(mt * 16 + r + 8 * half) * DV + nt * 16 + nlow] = c[r];
  }
  if (tid < DK) {
    const float* skf = sk + (size_t)bh * 65536;
    float acc = 0.f;
    for (int l = 0; l < LSEG; ++l) acc += skf[(size_t)l * DK + tid];
    z[bh * DK + tid] += acc;
  }
}

// ---------------- causal local attention + gated combine ----------------
__global__ void attn_kernel(const __bf16* __restrict__ qbf, const __bf16* __restrict__ kbf,
                            const __bf16* __restrict__ vT, const float* __restrict__ amem,
                            const float* __restrict__ beta, int seg, float* __restrict__ out) {
  extern __shared__ float smem[];
  int bh = blockIdx.y;
  int wave = threadIdx.x >> 5;
  int lt = blockIdx.x * 4 + wave;      // 16-row tile of L
  int lane = threadIdx.x & 31;
  int half = lane >> 4, nlow = lane & 15;
  int b = bh >> 3, h = bh & 7;
  float* S = smem + (size_t)wave * 16 * SPAD;
  const __bf16* qb = qbf + (size_t)bh * 65536;
  const __bf16* kb = kbf + (size_t)bh * 65536;
  const __bf16* vb = vT  + (size_t)bh * 65536;
  const float inv_sqrt = 0.088388347648318447f;   // 1/sqrt(128)

  // preload q A-fragments (K=128 in 4 chunks of 32)
  v16bf qa[4];
  {
    const __bf16* qrow = qb + (size_t)(lt * 16 + nlow) * DK;
#pragma unroll
    for (int kbk = 0; kbk < 4; ++kbk) {
      V16U a;
      int k0 = kbk * 32 + half * 8;
      a.h[0] = *(const bf16x8*)(qrow + k0);
      a.h[1] = *(const bf16x8*)(qrow + k0 + 16);
      qa[kbk] = a.v;
    }
  }
  // scores -> LDS (only causal column tiles)
  for (int ct = 0; ct <= lt; ++ct) {
    v8f c = zero8();
    const __bf16* krow = kb + (size_t)(ct * 16 + nlow) * DK;
#pragma unroll
    for (int kbk = 0; kbk < 4; ++kbk) {
      V16U bb;
      int k0 = kbk * 32 + half * 8;
      bb.h[0] = *(const bf16x8*)(krow + k0);
      bb.h[1] = *(const bf16x8*)(krow + k0 + 16);
      c = __builtin_amdgcn_wmma_f32_16x16x32_bf16(false, qa[kbk], false, bb.v, (short)0, c, false, false);
    }
#pragma unroll
    for (int r = 0; r < 8; ++r) {
      int m = r + 8 * half;
      float sv = c[r] * inv_sqrt;
      if (ct == lt && nlow > m) sv = -3.0e38f;     // causal mask on diagonal tile
      S[m * SPAD + ct * 16 + nlow] = sv;
    }
  }
  __syncthreads();

  int ncols = (lt + 1) * 16;
  // row stats for row nlow (lane pair {L, L+16} splits the scan, xor-16 combines)
  float mx = -3.0e38f;
  {
    const float* row = S + nlow * SPAD;
    for (int cidx = half; cidx < ncols; cidx += 2) mx = fmaxf(mx, row[cidx]);
    mx = fmaxf(mx, __shfl_xor(mx, 16));
  }
  float sum = 0.f;
  {
    const float* row = S + nlow * SPAD;
    for (int cidx = half; cidx < ncols; cidx += 2) sum += __expf(row[cidx] - mx);
    sum += __shfl_xor(sum, 16);
  }
  float inv_sum = 1.f / sum;

  // a_dot = P @ V (P fragments built from LDS on the fly)
  v8f o[8];
#pragma unroll
  for (int t = 0; t < 8; ++t) o[t] = zero8();
  for (int cc = 0; cc * 32 < ncols; ++cc) {
    V16U pa;
#pragma unroll
    for (int e = 0; e < 16; ++e) {
      int lp = cc * 32 + (e >> 3) * 16 + half * 8 + (e & 7);
      float p = 0.f;
      if (lp < ncols) p = __expf(S[nlow * SPAD + lp] - mx) * inv_sum;
      pa.v[e] = (__bf16)p;
    }
#pragma unroll
    for (int et = 0; et < 8; ++et) {
      V16U bb;
      const __bf16* vrow = vb + (size_t)(et * 16 + nlow) * LSEG + cc * 32 + half * 8;
      bb.h[0] = *(const bf16x8*)(vrow);
      bb.h[1] = *(const bf16x8*)(vrow + 16);
      o[et] = __builtin_amdgcn_wmma_f32_16x16x32_bf16(false, pa.v, false, bb.v, (short)0, o[et], false, false);
    }
  }
  // gated combine + write
  float g = 1.f / (1.f + __expf(-beta[h]));
  const float* am = amem + (size_t)bh * 65536;
#pragma unroll
  for (int r = 0; r < 8; ++r) {
    int m = r + 8 * half;
    int l = lt * 16 + m;
    size_t orow = (size_t)(b * SEQ + seg * LSEG + l) * NQK + h * DK;
#pragma unroll
    for (int et = 0; et < 8; ++et) {
      int e = et * 16 + nlow;
      out[orow + e] = g * am[(size_t)l * DK + e] + (1.f - g) * o[et][r];
    }
  }
}

// ---------------- host launcher ----------------
extern "C" void kernel_launch(void* const* d_in, const int* in_sizes, int n_in,
                              void* d_out, int out_size, void* d_ws, size_t ws_size,
                              hipStream_t stream) {
  const float* x    = (const float*)d_in[0];
  const float* Wq   = (const float*)d_in[1];
  const float* bq   = (const float*)d_in[2];
  const float* Wk   = (const float*)d_in[3];
  const float* bk   = (const float*)d_in[4];
  const float* Wv   = (const float*)d_in[5];
  const float* bv   = (const float*)d_in[6];
  const float* beta = (const float*)d_in[7];
  float* out = (float*)d_out;
  (void)in_sizes; (void)n_in; (void)out_size; (void)ws_size;

  char* ws = (char*)d_ws;
  size_t off = 0;
  auto alloc = [&](size_t bytes) -> char* {
    char* p = ws + off;
    off += (bytes + 255) & ~(size_t)255;
    return p;
  };
  __bf16* xbf = (__bf16*)alloc((size_t)NROWS * DIN * 2);
  __bf16* wt  = (__bf16*)alloc((size_t)3 * DIN * NQK * 2);
  float*  Qb  = (float*)alloc((size_t)NROWS * NQK * 4);
  float*  Kb  = (float*)alloc((size_t)NROWS * NQK * 4);
  float*  Vb  = (float*)alloc((size_t)NROWS * NQK * 4);
  float*  sq  = (float*)alloc((size_t)16 * LSEG * DK * 4);
  float*  sk  = (float*)alloc((size_t)16 * LSEG * DK * 4);
  __bf16* qbf = (__bf16*)alloc((size_t)16 * LSEG * DK * 2);
  __bf16* kbf = (__bf16*)alloc((size_t)16 * LSEG * DK * 2);
  __bf16* vT  = (__bf16*)alloc((size_t)16 * DK * LSEG * 2);
  __bf16* skT = (__bf16*)alloc((size_t)16 * DK * LSEG * 2);
  __bf16* vdT = (__bf16*)alloc((size_t)16 * DK * LSEG * 2);
  float* amem = (float*)alloc((size_t)16 * LSEG * DK * 4);
  float* Mst  = (float*)alloc((size_t)16 * DK * DV * 4);
  float* zst  = (float*)alloc((size_t)16 * DK * 4);

  hipMemsetAsync(Mst, 0, (size_t)16 * DK * DV * 4, stream);
  hipMemsetAsync(zst, 0, (size_t)16 * DK * 4, stream);

  conv_x_kernel<<<(NROWS * DIN) / 256, 256, 0, stream>>>(x, xbf, NROWS * DIN);
  conv_w_kernel<<<dim3((DIN * NQK) / 256, 1, 3), 256, 0, stream>>>(Wq, Wk, Wv, wt);
  proj_kernel<<<dim3(NQK / 32, NROWS / 64, 3), 32, 0, stream>>>(xbf, wt, bq, bk, bv, Qb, Kb, Vb);

  for (int seg = 0; seg < NSEG; ++seg) {
    phi_kernel<<<(16 * LSEG * DK) / 256, 256, 0, stream>>>(Qb, Kb, Vb, seg, sq, sk, qbf, kbf, vT, skT);
    retrieve_kernel<<<dim3(LSEG / 64, 16), 128, 0, stream>>>(sq, sk, Vb, Mst, zst, seg, amem, vdT);
    update_kernel<<<16, 512, 262144, stream>>>(skT, vdT, sk, Mst, zst);
    attn_kernel<<<dim3(8, 16), 128, 4 * 16 * SPAD * sizeof(float), stream>>>(qbf, kbf, vT, amem, beta, seg, out);
  }
}
